// MultiHeadAttention_55783035240901
// MI455X (gfx1250) — matrix operand order repaired and checked by reading, not compiled, not run on hardware
//
#include <hip/hip_runtime.h>

// ---------------------------------------------------------------------------
// MHA + RoPE for MI455X (gfx1250, wave32, WMMA bf16 -> f32)
//   stage 1: fp32 -> bf16 converts
//   stage 2: WMMA GEMM x*W^T for Q,K,V (epilogue scatters to [B,H,S,Hd] / V^T)
//   stage 3: in-place RoPE on Q,K
//   stage 4: causal flash attention, async-LDS double-buffered K/V tiles
//   stage 5: WMMA GEMM O*Wo^T -> fp32 output
// ---------------------------------------------------------------------------

typedef __bf16 bf16_t;
typedef __attribute__((ext_vector_type(16))) __bf16 v16bf;
typedef __attribute__((ext_vector_type(8)))  __bf16 v8bf;
typedef __attribute__((ext_vector_type(8)))  float  v8f;
typedef int v4i __attribute__((vector_size(16)));
typedef __attribute__((address_space(1))) v4i* gv4i_p;   // global int4*
typedef __attribute__((address_space(3))) v4i* lv4i_p;   // LDS int4*

#define SEQ   2048
#define DIM   1024
#define NH    16
#define HD    64
#define BATCH 2
#define BH    (BATCH*NH)     // 32
#define MROWS (BATCH*SEQ)    // 4096

#if __has_builtin(__builtin_amdgcn_global_load_async_to_lds_b128)
#define HAVE_ASYNC_LDS 1
#else
#define HAVE_ASYNC_LDS 0
#endif

// ---------------- small helpers -------------------------------------------

__device__ __forceinline__ bf16_t f2bf(float x) {
    union { unsigned short u; bf16_t b; } cv;
    unsigned u32 = __float_as_uint(x);
    cv.u = (unsigned short)((u32 + 0x7fffu + ((u32 >> 16) & 1u)) >> 16); // RNE
    return cv.b;
}
__device__ __forceinline__ float bf2f(bf16_t b) {
    union { bf16_t b; unsigned short u; } cv; cv.b = b;
    return __uint_as_float((unsigned)cv.u << 16);
}

__device__ __forceinline__ v8f wmma_bf16(v16bf a, v16bf b, v8f c) {
    return __builtin_amdgcn_wmma_f32_16x16x32_bf16(
        false, a, false, b, (short)0, c, false, false);
}

// A-fragment (16x32, 16-bit): lane<16 -> row=lane, K {k0..k0+7, k0+16..k0+23}
//                             lane>=16 -> row=lane-16, K {k0+8..15, k0+24..31}
__device__ __forceinline__ v16bf load_a16(const bf16_t* base, int ld,
                                          int m0, int k0, int lane) {
    int row = m0 + (lane & 15);
    int ko  = k0 + ((lane >> 4) << 3);
    const bf16_t* p = base + (size_t)row * ld + ko;
    v8bf lo = *(const v8bf*)(p);
    v8bf hi = *(const v8bf*)(p + 16);
    v16bf r;
#pragma unroll
    for (int i = 0; i < 8; ++i) { r[i] = lo[i]; r[i + 8] = hi[i]; }
    return r;
}

// B-fragment (32x16) where B[k][n] = Src[n0+n][k0+k] (Src rows K-contiguous):
// lane n holds column n, K {k0..k0+7, k0+16..k0+23} ; lane n+16 holds K {k0+8..15, k0+24..31}.
__device__ __forceinline__ v16bf load_bT16(const bf16_t* base, int ld,
                                           int n0, int k0, int lane) {
    int row = n0 + (lane & 15);
    int ko  = k0 + ((lane >> 4) << 3);
    const bf16_t* p = base + (size_t)row * ld + ko;
    v8bf lo = *(const v8bf*)(p);
    v8bf hi = *(const v8bf*)(p + 16);
    v16bf r;
#pragma unroll
    for (int i = 0; i < 8; ++i) { r[i] = lo[i]; r[i + 8] = hi[i]; }
    return r;
}

#if HAVE_ASYNC_LDS
// Async-copy a fully contiguous 4KB block to LDS (8 issues x 32 lanes x 16B).
__device__ __forceinline__ void async_copy_contig(const bf16_t* g, bf16_t* l,
                                                  int lane) {
#pragma unroll
    for (int i = 0; i < 8; ++i) {
        int off = (i * 32 + lane) * 16;
        __builtin_amdgcn_global_load_async_to_lds_b128(
            (gv4i_p)((const char*)g + off),
            (lv4i_p)((char*)l + off), 0, 0);
    }
}
// Async-copy 64 rows x 64 bytes (V^T tile; row stride = gld elements).
__device__ __forceinline__ void async_copy_rows(const bf16_t* g, int gld,
                                                bf16_t* l, int lane) {
#pragma unroll
    for (int i = 0; i < 8; ++i) {
        int c = i * 32 + lane;                  // chunk 0..255, 16B each
        int row = c >> 2, part = c & 3;
        __builtin_amdgcn_global_load_async_to_lds_b128(
            (gv4i_p)((const char*)(g + (size_t)row * gld) + part * 16),
            (lv4i_p)((char*)l + c * 16), 0, 0);
    }
}
#endif

// ---------------- stage 1: fp32 -> bf16 -----------------------------------

__global__ void conv_bf16_kernel(const float* __restrict__ src,
                                 bf16_t* __restrict__ dst, int n) {
    int i = blockIdx.x * blockDim.x + threadIdx.x;
    if (i < n) dst[i] = f2bf(src[i]);
}

// ---------------- stage 2/5: GEMM  C = A(MxK) * W(NxK)^T ------------------
// wave register block 32x64: 2 A-frags x 4 B-frags -> 8 WMMA per K-step.
// mode 0: f32 row-major [M,N]; mode 1: bf16 [B,H,S,Hd]; mode 2: bf16 [B,H,Hd,S]

__global__ void __launch_bounds__(128)
gemm_bf16_kernel(const bf16_t* __restrict__ A, const bf16_t* __restrict__ W,
                 float* __restrict__ Cf, bf16_t* __restrict__ Cb, int mode) {
    const int wave = threadIdx.x >> 5;
    const int lane = threadIdx.x & 31;
    const int m0 = blockIdx.y * 128 + wave * 32;
    const int n0 = blockIdx.x * 64;

    v8f c[2][4] = {};
    for (int k0 = 0; k0 < DIM; k0 += 32) {
        if (k0 + 64 < DIM) {  // uniform branch; warm caches two K-steps ahead
            __builtin_prefetch(A + (size_t)(m0 + (lane & 31)) * DIM + k0 + 64, 0, 0);
            __builtin_prefetch(W + (size_t)(n0 + (lane & 31)) * DIM + k0 + 64, 0, 0);
        }
        v16bf a0 = load_a16(A, DIM, m0,      k0, lane);
        v16bf a1 = load_a16(A, DIM, m0 + 16, k0, lane);
#pragma unroll
        for (int t = 0; t < 4; ++t) {
            v16bf b = load_bT16(W, DIM, n0 + t * 16, k0, lane);
            c[0][t] = wmma_bf16(a0, b, c[0][t]);
            c[1][t] = wmma_bf16(a1, b, c[1][t]);
        }
    }

    const int ncol = lane & 15;
    const int mofs = (lane >> 4) << 3;
#pragma unroll
    for (int mt = 0; mt < 2; ++mt) {
#pragma unroll
        for (int t = 0; t < 4; ++t) {
#pragma unroll
            for (int r = 0; r < 8; ++r) {
                int m = m0 + mt * 16 + mofs + r;
                int n = n0 + t * 16 + ncol;
                if (mode == 0) {
                    Cf[(size_t)m * DIM + n] = c[mt][t][r];
                } else {
                    int b = m >> 11, s = m & (SEQ - 1);
                    int h = n >> 6,  d = n & (HD - 1);
                    if (mode == 1)
                        Cb[(((size_t)(b * NH + h) * SEQ + s) << 6) + d] = f2bf(c[mt][t][r]);
                    else
                        Cb[((size_t)(b * NH + h) * HD + d) * SEQ + s]   = f2bf(c[mt][t][r]);
                }
            }
        }
    }
}

// ---------------- stage 3: in-place RoPE on Q and K -----------------------

__global__ void rope_kernel(bf16_t* __restrict__ Qb, bf16_t* __restrict__ Kb) {
    int idx = blockIdx.x * blockDim.x + threadIdx.x;   // BH*SEQ*32 threads
    int j  = idx & 31;
    int s  = (idx >> 5) & (SEQ - 1);
    int bh = idx >> 16;
    if (bh >= BH) return;
    size_t base = ((size_t)bh * SEQ + s) * HD;
    float f   = __expf(-0.2878231366f * (float)j);     // 10000^(-2j/64)
    float ang = (float)s * f;
    float cs = __cosf(ang), sn = __sinf(ang);

    float q1 = bf2f(Qb[base + j]), q2 = bf2f(Qb[base + j + 32]);
    Qb[base + j]      = f2bf(q1 * cs - q2 * sn);
    Qb[base + j + 32] = f2bf(q2 * cs + q1 * sn);

    float k1 = bf2f(Kb[base + j]), k2 = bf2f(Kb[base + j + 32]);
    Kb[base + j]      = f2bf(k1 * cs - k2 * sn);
    Kb[base + j + 32] = f2bf(k2 * cs + k1 * sn);
}

// ---------------- stage 4: causal flash attention -------------------------
// 1 wave per 16-row Q tile, 32-key steps. K/V tiles double-buffered in LDS
// via GLOBAL_LOAD_ASYNC_TO_LDS_B128 (prefetch kt+1 while computing kt),
// drained with s_wait_asynccnt. Falls back to direct global loads if the
// async builtin is unavailable.

__global__ void __launch_bounds__(128)
flash_attn_kernel(const bf16_t* __restrict__ Q, const bf16_t* __restrict__ K,
                  const bf16_t* __restrict__ Vt, bf16_t* __restrict__ O) {
    __shared__ bf16_t pbuf[4][16 * 32];
#if HAVE_ASYNC_LDS
    __shared__ bf16_t kst[4][2][32 * HD];   // 32 keys x 64 d  (4KB per buf)
    __shared__ bf16_t vst[4][2][HD * 32];   // 64 d    x 32 s  (4KB per buf)
#endif

    const int wave = threadIdx.x >> 5;
    const int lane = threadIdx.x & 31;
    const int qblk = blockIdx.x & 31;        // SEQ/64 = 32
    const int bh   = blockIdx.x >> 5;
    const int q0   = qblk * 64 + wave * 16;

    const bf16_t* Qp = Q  + (size_t)bh * SEQ * HD;
    const bf16_t* Kp = K  + (size_t)bh * SEQ * HD;
    const bf16_t* Vp = Vt + (size_t)bh * HD * SEQ;
    bf16_t* myp = pbuf[wave];

    const int ncol = lane & 15;
    const int mofs = (lane >> 4) << 3;

    v16bf qf0 = load_a16(Qp, HD, q0, 0,  lane);
    v16bf qf1 = load_a16(Qp, HD, q0, 32, lane);

    float m_r[8], l_r[8];
    v8f o[4] = {};
#pragma unroll
    for (int r = 0; r < 8; ++r) { m_r[r] = -1e30f; l_r[r] = 0.0f; }

    const int nt = (q0 + 15) / 32 + 1;       // causal tile count

#if HAVE_ASYNC_LDS
    async_copy_contig(Kp, kst[wave][0], lane);           // K tile 0 (contig 4KB)
    async_copy_rows(Vp, SEQ, vst[wave][0], lane);        // V^T tile 0
#endif

    for (int kt = 0; kt < nt; ++kt) {
        const int k0 = kt * 32;
#if HAVE_ASYNC_LDS
        const int cur = kt & 1;
        if (kt + 1 < nt) {
            async_copy_contig(Kp + (size_t)(k0 + 32) * HD, kst[wave][cur ^ 1], lane);
            async_copy_rows(Vp + k0 + 32, SEQ, vst[wave][cur ^ 1], lane);
            asm volatile("s_wait_asynccnt 16" ::: "memory");  // tile kt done
        } else {
            asm volatile("s_wait_asynccnt 0" ::: "memory");
        }
        const bf16_t* ksrc = kst[wave][cur];  const int kld = HD;
        const bf16_t* vsrc = vst[wave][cur];  const int vld = 32;
#else
        const bf16_t* ksrc = Kp + (size_t)k0 * HD;  const int kld = HD;
        const bf16_t* vsrc = Vp + k0;               const int vld = SEQ;
#endif

        // ---- scores S = (Q * K^T) * Hd^-0.5, two 16-key sub-tiles --------
        v8f sc[2];
#pragma unroll
        for (int t = 0; t < 2; ++t) {
            v16bf kb0 = load_bT16(ksrc, kld, t * 16, 0,  lane);
            v16bf kb1 = load_bT16(ksrc, kld, t * 16, 32, lane);
            v8f acc = {};
            acc = wmma_bf16(qf0, kb0, acc);
            acc = wmma_bf16(qf1, kb1, acc);
            sc[t] = acc;
        }

        // ---- causal mask + online softmax --------------------------------
#pragma unroll
        for (int r = 0; r < 8; ++r) {
            int qrow = q0 + mofs + r;
            float s0 = sc[0][r] * 0.125f;
            float s1 = sc[1][r] * 0.125f;
            if (k0 + ncol      > qrow) s0 = -1e30f;
            if (k0 + 16 + ncol > qrow) s1 = -1e30f;

            float rm = fmaxf(s0, s1);
#pragma unroll
            for (int msk = 1; msk < 16; msk <<= 1)
                rm = fmaxf(rm, __shfl_xor(rm, msk, 32));
            float mnew = fmaxf(m_r[r], rm);
            float corr = __expf(m_r[r] - mnew);
            float p0 = __expf(s0 - mnew);
            float p1 = __expf(s1 - mnew);
            float ps = p0 + p1;
#pragma unroll
            for (int msk = 1; msk < 16; msk <<= 1)
                ps += __shfl_xor(ps, msk, 32);
            l_r[r] = l_r[r] * corr + ps;
            m_r[r] = mnew;
#pragma unroll
            for (int t = 0; t < 4; ++t) o[t][r] *= corr;

            myp[(mofs + r) * 32 + ncol]      = f2bf(p0);
            myp[(mofs + r) * 32 + 16 + ncol] = f2bf(p1);
        }

        asm volatile("s_wait_dscnt 0" ::: "memory");   // same-wave LDS in-order

        v16bf pa = load_a16(myp, 32, 0, 0, lane);
#pragma unroll
        for (int t = 0; t < 4; ++t) {
            v16bf vb = load_bT16(vsrc, vld, t * 16, 0, lane);
            o[t] = wmma_bf16(pa, vb, o[t]);
        }
    }

    // ---- normalize and store into [B,S,D] bf16 ---------------------------
    const int b = bh >> 4, h = bh & 15;
#pragma unroll
    for (int r = 0; r < 8; ++r) {
        float inv = 1.0f / l_r[r];
        size_t row = ((size_t)b * SEQ + q0 + mofs + r) * DIM + h * HD;
#pragma unroll
        for (int t = 0; t < 4; ++t)
            O[row + t * 16 + ncol] = f2bf(o[t][r] * inv);
    }
}

// ---------------- host-side orchestration ---------------------------------

extern "C" void kernel_launch(void* const* d_in, const int* in_sizes, int n_in,
                              void* d_out, int out_size, void* d_ws, size_t ws_size,
                              hipStream_t stream) {
    (void)in_sizes; (void)n_in; (void)out_size; (void)ws_size;

    const float* x  = (const float*)d_in[0];
    const float* Wq = (const float*)d_in[1];
    const float* Wk = (const float*)d_in[2];
    const float* Wv = (const float*)d_in[3];
    const float* Wo = (const float*)d_in[4];
    float* out = (float*)d_out;

    char* ws = (char*)d_ws;
    const size_t MB = 1u << 20;
    bf16_t* xb  = (bf16_t*)(ws + 0 * MB);    // 8 MiB  [4096,1024]
    bf16_t* wqb = (bf16_t*)(ws + 8 * MB);    // 2 MiB
    bf16_t* wkb = (bf16_t*)(ws + 10 * MB);   // 2 MiB
    bf16_t* wvb = (bf16_t*)(ws + 12 * MB);   // 2 MiB
    bf16_t* wob = (bf16_t*)(ws + 14 * MB);   // 2 MiB
    bf16_t* Qb  = (bf16_t*)(ws + 16 * MB);   // 8 MiB  [B,H,S,Hd]
    bf16_t* Kb  = (bf16_t*)(ws + 24 * MB);   // 8 MiB  [B,H,S,Hd]
    bf16_t* Vtb = (bf16_t*)(ws + 32 * MB);   // 8 MiB  [B,H,Hd,S]
    bf16_t* Ob  = (bf16_t*)(ws + 40 * MB);   // 8 MiB  [B,S,D]

    const int NX = MROWS * DIM;   // 4194304
    const int NW = DIM * DIM;     // 1048576
    conv_bf16_kernel<<<(NX + 255) / 256, 256, 0, stream>>>(x,  xb,  NX);
    conv_bf16_kernel<<<(NW + 255) / 256, 256, 0, stream>>>(Wq, wqb, NW);
    conv_bf16_kernel<<<(NW + 255) / 256, 256, 0, stream>>>(Wk, wkb, NW);
    conv_bf16_kernel<<<(NW + 255) / 256, 256, 0, stream>>>(Wv, wvb, NW);
    conv_bf16_kernel<<<(NW + 255) / 256, 256, 0, stream>>>(Wo, wob, NW);

    dim3 ggrid(DIM / 64, MROWS / 128);       // (16, 32)
    gemm_bf16_kernel<<<ggrid, 128, 0, stream>>>(xb, wqb, nullptr, Qb,  1);
    gemm_bf16_kernel<<<ggrid, 128, 0, stream>>>(xb, wkb, nullptr, Kb,  1);
    gemm_bf16_kernel<<<ggrid, 128, 0, stream>>>(xb, wvb, nullptr, Vtb, 2);

    const int NR = BH * SEQ * 32;            // 2097152 rotation pairs
    rope_kernel<<<NR / 256, 256, 0, stream>>>(Qb, Kb);

    flash_attn_kernel<<<BH * (SEQ / 64), 128, 0, stream>>>(Qb, Kb, Vtb, Ob);

    gemm_bf16_kernel<<<ggrid, 128, 0, stream>>>(Ob, wob, out, nullptr, 0);
}
